// TSLTM_979252543717
// MI455X (gfx1250) — matrix operand order repaired and checked by reading, not compiled, not run on hardware
//
#include <hip/hip_runtime.h>
#include <hip/hip_bf16.h>
#include <stdint.h>

// ---------------------------------------------------------------------------
// TSLTM for MI455X (gfx1250, wave32, WMMA bf16)
// B=64, T=512, D=256, H=256
// ---------------------------------------------------------------------------
constexpr int Bc = 64;
constexpr int Tc = 512;
constexpr int Dc = 256;
constexpr int Hc = 256;
constexpr int G4 = 4 * Hc;    // 1024
constexpr int KT = Dc + Hc;   // 512  (stacked [W;U] K dimension)
constexpr int LS = 264;       // padded LDS row stride (bf16 elems): 528B = 132 dw -> bank-friendly

typedef __attribute__((ext_vector_type(16))) __bf16 v16bf;
typedef __attribute__((ext_vector_type(8)))  float  v8f;

struct P128x2 { uint4 lo, hi; };   // 32 bytes == one v16bf fragment

static __device__ __forceinline__ v16bf mkfrag(uint4 a, uint4 b) {
    P128x2 p{a, b};
    return __builtin_bit_cast(v16bf, p);
}

static __device__ __forceinline__ unsigned short f2bf(float f) {
    unsigned int u = __builtin_bit_cast(unsigned int, f);
    u += 0x7fffu + ((u >> 16) & 1u);           // round-to-nearest-even
    return (unsigned short)(u >> 16);
}

// ---- fast activations on native gfx1250 transcendentals -------------------
static __device__ __forceinline__ float fexp(float x) {          // e^x
    return __builtin_amdgcn_exp2f(x * 1.4426950408889634f);
}
static __device__ __forceinline__ float frcp(float x) {
    return __builtin_amdgcn_rcpf(x);
}
static __device__ __forceinline__ float fsigmoid(float x) {
    return frcp(1.0f + fexp(-x));
}
static __device__ __forceinline__ float ftanh(float x) {
#if __has_builtin(__builtin_amdgcn_tanhf)
    return __builtin_amdgcn_tanhf(x);                            // v_tanh_f32
#else
    float e = fexp(2.0f * x);
    return 1.0f - 2.0f * frcp(e + 1.0f);
#endif
}

// ---------------------------------------------------------------------------
// Prep: pack W,U into bf16 column-major [n][k] (k = 0..255 -> W, 256..511 -> U)
// and W_d into bf16 column-major [n][k].  Lives in d_ws (L2-resident, 1.13 MB).
// ---------------------------------------------------------------------------
__global__ void tsltm_pack_weights(const float* __restrict__ W,
                                   const float* __restrict__ U,
                                   const float* __restrict__ Wd,
                                   unsigned short* __restrict__ wu_b,
                                   unsigned short* __restrict__ wd_b) {
    int idx = blockIdx.x * blockDim.x + threadIdx.x;
    const int NWU = G4 * KT;          // 524288
    const int NWD = Hc * Hc;          // 65536
    if (idx < NWU) {
        int n = idx >> 9;             // /512
        int k = idx & (KT - 1);
        float v = (k < Dc) ? W[(size_t)k * G4 + n] : U[(size_t)(k - Dc) * G4 + n];
        wu_b[(size_t)n * KT + k] = f2bf(v);
    } else if (idx < NWU + NWD) {
        int j = idx - NWU;
        int n = j >> 8;
        int k = j & (Hc - 1);
        wd_b[(size_t)n * Hc + k] = f2bf(Wd[(size_t)k * Hc + n]);
    }
}

// ---------------------------------------------------------------------------
// Persistent scan kernel: 4 blocks x 512 threads (16 waves).
// Block handles 16 batch rows; wave w owns H-columns [16w, 16w+16).
// Per step per wave: 4 gate tiles (K=512) + 1 c_s tile (K=256) = 72 WMMAs.
// Weights stream from L2 every step; an opaque per-iteration *offset* (not a
// type-erased pointer!) defeats LICM hoisting/spilling while preserving
// global address-space provenance -> global_load_b128, not flat_load.
// ---------------------------------------------------------------------------
__global__ __launch_bounds__(512, 1)
void tsltm_scan(const float* __restrict__ x,      // [B,T,D]
                const float* __restrict__ td,     // [B,T]
                const float* __restrict__ bias,   // [4H]
                const unsigned short* __restrict__ wu_in,  // [G4][KT] bf16, col-major
                const unsigned short* __restrict__ wd_in,  // [H][H]   bf16, col-major
                float* __restrict__ hseq,         // [B,T,H]
                float* __restrict__ hf,           // [B,H]
                float* __restrict__ cf) {         // [B,H]
    const int tid  = threadIdx.x;
    const int w    = tid >> 5;          // wave id 0..15
    const int l    = tid & 31;          // lane
    const int half = l >> 4;            // 0/1
    const int cN   = l & 15;            // fragment column (N) / A row (M)
    const int bm0  = blockIdx.x * 16;   // first batch row of this block
    const int hcol = w * 16;            // first H-column owned by this wave
    const float E1 = 2.718281828459045f;

    __shared__ alignas(16) unsigned short sx[16 * LS];
    __shared__ alignas(16) unsigned short sh[16 * LS];
    __shared__ alignas(16) unsigned short sc[16 * LS];
    __shared__ float srdt[16];

    // zero h0 / c0
    for (int i = tid; i < 16 * LS; i += 512) { sh[i] = 0; sc[i] = 0; }

    // per-lane bias for the 4 gate columns this lane produces
    float bg[4];
    #pragma unroll
    for (int g = 0; g < 4; ++g) bg[g] = bias[g * Hc + hcol + cN];

    v8f cfr = {};     // f32 cell state, persistent in registers
    v8f hlast = {};

    #pragma clang loop unroll(disable)
    for (int t = 0; t < Tc; ++t) {
        // Opaque zero offset: loop-variant to LLVM, so weight loads can't be
        // hoisted out of the 512-step loop; pointer provenance preserved.
        unsigned int zoff = 0;
        asm volatile("" : "+s"(zoff));
        const unsigned short* wu = wu_in + zoff;
        const unsigned short* wd = wd_in + zoff;

        // ---- stage x_t (16 x 256 f32 -> bf16, vectorized x4) --------------
        for (int e = tid; e < 16 * (Dc / 4); e += 512) {
            int r = e >> 6, c4 = (e & 63) * 4;
            const float4 xv = *reinterpret_cast<const float4*>(
                &x[(size_t)(bm0 + r) * Tc * Dc + (size_t)t * Dc + c4]);
            ushort4 o;
            o.x = f2bf(xv.x); o.y = f2bf(xv.y); o.z = f2bf(xv.z); o.w = f2bf(xv.w);
            *reinterpret_cast<ushort4*>(&sx[r * LS + c4]) = o;
        }
        if (tid < 16) {
            float dt = td[(size_t)(bm0 + tid) * Tc + t];
            // 1/ln(e+dt) = rcp(ln2 * log2(e+dt))
            srdt[tid] = frcp(0.6931471805599453f * __builtin_amdgcn_logf(E1 + dt));
        }
        __syncthreads();

        // ---- gates = [x_t, h_t] @ [W;U]  (K = 512, bf16 WMMA, f32 acc) ----
        v8f a0 = {}, a1 = {}, a2 = {}, a3 = {};
        #pragma unroll
        for (int k = 0; k < 16; ++k) {
            const unsigned short* abase = (k < 8) ? sx : sh;
            const int kk = (k < 8) ? k : (k - 8);
            const uint4* ap = reinterpret_cast<const uint4*>(
                abase + cN * LS + kk * 32 + half * 8);
            v16bf a = mkfrag(ap[0], ap[2]);   // K gap of 16 elems between halves

            const size_t kb = (size_t)k * 32 + half * 8;
            const uint4* b0 = reinterpret_cast<const uint4*>(wu + (size_t)(0 * Hc + hcol + cN) * KT + kb);
            const uint4* b1 = reinterpret_cast<const uint4*>(wu + (size_t)(1 * Hc + hcol + cN) * KT + kb);
            const uint4* b2 = reinterpret_cast<const uint4*>(wu + (size_t)(2 * Hc + hcol + cN) * KT + kb);
            const uint4* b3 = reinterpret_cast<const uint4*>(wu + (size_t)(3 * Hc + hcol + cN) * KT + kb);
            a0 = __builtin_amdgcn_wmma_f32_16x16x32_bf16(false, a, false, mkfrag(b0[0], b0[2]), (short)0, a0, false, false);
            a1 = __builtin_amdgcn_wmma_f32_16x16x32_bf16(false, a, false, mkfrag(b1[0], b1[2]), (short)0, a1, false, false);
            a2 = __builtin_amdgcn_wmma_f32_16x16x32_bf16(false, a, false, mkfrag(b2[0], b2[2]), (short)0, a2, false, false);
            a3 = __builtin_amdgcn_wmma_f32_16x16x32_bf16(false, a, false, mkfrag(b3[0], b3[2]), (short)0, a3, false, false);
        }

        // ---- c_s = c_t @ W_d  (K = 256, B streamed from L2) ----
        v8f cs = {};
        #pragma unroll
        for (int k = 0; k < 8; ++k) {
            const uint4* ap = reinterpret_cast<const uint4*>(
                sc + cN * LS + k * 32 + half * 8);
            v16bf a = mkfrag(ap[0], ap[2]);
            const uint4* bp = reinterpret_cast<const uint4*>(
                wd + (size_t)(hcol + cN) * Hc + k * 32 + half * 8);
            cs = __builtin_amdgcn_wmma_f32_16x16x32_bf16(false, a, false, mkfrag(bp[0], bp[2]), (short)0, cs, false, false);
        }

        // ---- elementwise LSTM math on the 16x16 f32 fragment ----
        v8f hn;
        #pragma unroll
        for (int v = 0; v < 8; ++v) {
            const int rm = v + half * 8;                 // batch row within block
            float csv  = ftanh(cs[v]);
            float cadj = (cfr[v] - csv) + csv * srdt[rm];
            float gi = fsigmoid(a0[v] + bg[0]);
            float gf = fsigmoid(a1[v] + bg[1]);
            float gg = ftanh  (a2[v] + bg[2]);
            float go = fsigmoid(a3[v] + bg[3]);
            float cn = gf * cadj + gi * gg;
            float h  = go * ftanh(cn);
            cfr[v] = cn;
            hn[v]  = h;
            hseq[(size_t)(bm0 + rm) * Tc * Hc + (size_t)t * Hc + hcol + cN] = h;
        }
        hlast = hn;

        __syncthreads();   // all LDS reads of sx/sh/sc done before rewrite

        // ---- publish new h, c as bf16 A-operands for step t+1 ----
        #pragma unroll
        for (int v = 0; v < 8; ++v) {
            const int rm = v + half * 8;
            sh[rm * LS + hcol + cN] = f2bf(hn[v]);
            sc[rm * LS + hcol + cN] = f2bf(cfr[v]);
        }
    }

    // ---- final states ----
    #pragma unroll
    for (int v = 0; v < 8; ++v) {
        const int rm = v + half * 8;
        hf[(size_t)(bm0 + rm) * Hc + hcol + cN] = hlast[v];
        cf[(size_t)(bm0 + rm) * Hc + hcol + cN] = cfr[v];
    }
}

// ---------------------------------------------------------------------------
extern "C" void kernel_launch(void* const* d_in, const int* in_sizes, int n_in,
                              void* d_out, int out_size, void* d_ws, size_t ws_size,
                              hipStream_t stream) {
    const float* x    = (const float*)d_in[0];   // [B,T,D]
    const float* td   = (const float*)d_in[1];   // [B,T]
    const float* Wd   = (const float*)d_in[2];   // [H,H]
    const float* W    = (const float*)d_in[3];   // [D,4H]
    const float* U    = (const float*)d_in[4];   // [H,4H]
    const float* bias = (const float*)d_in[5];   // [4H]

    unsigned short* wu_b = (unsigned short*)d_ws;                 // 1024*512 bf16
    unsigned short* wd_b = wu_b + (size_t)G4 * KT;                // 256*256 bf16

    float* hseq = (float*)d_out;                                  // [B,T,H]
    float* hf   = hseq + (size_t)Bc * Tc * Hc;                    // [B,H]
    float* cf   = hf + (size_t)Bc * Hc;                           // [B,H]

    {
        const int total = G4 * KT + Hc * Hc;
        const int thr = 256;
        tsltm_pack_weights<<<(total + thr - 1) / thr, thr, 0, stream>>>(
            W, U, Wd, wu_b, wd_b);
    }

    tsltm_scan<<<Bc / 16, 512, 0, stream>>>(x, td, bias, wu_b, wd_b, hseq, hf, cf);
}
